// Net_75771813036891
// MI455X (gfx1250) — matrix operand order repaired and checked by reading, not compiled, not run on hardware
//
#include <hip/hip_runtime.h>

typedef __attribute__((ext_vector_type(16))) _Float16 v16h;
typedef __attribute__((ext_vector_type(8)))  _Float16 v8h;
typedef __attribute__((ext_vector_type(8)))  float    v8f;

#define NSPINS 64
#define BATCH  16384
#define HCH    1280   // NSPINS * HID

#define BM 128
#define BK 32
#define LDT 48        // LDS row stride in halves (96 B, 16-B aligned)

// ---------------------------------------------------------------------------
// Apply autoregressive mask and convert f32 -> f16.
// mode 0: fc1 exclusive  (keep if c <  r%64)
// mode 1: fc2..fc5       (keep if c%64 <= r%64)
// mode 2: fc6            (keep if c%64 <= r)
// mode 3: plain convert  (sample)
// ---------------------------------------------------------------------------
__global__ void mask_cvt(const float* __restrict__ src, _Float16* __restrict__ dst,
                         int rows, int cols, int mode) {
    long long idx = (long long)blockIdx.x * blockDim.x + threadIdx.x;
    if (idx >= (long long)rows * cols) return;
    int r = (int)(idx / cols);
    int c = (int)(idx % cols);
    bool keep;
    if (mode == 0)      keep = c < (r & 63);
    else if (mode == 1) keep = (c & 63) <= (r & 63);
    else if (mode == 2) keep = (c & 63) <= r;
    else                keep = true;
    dst[idx] = keep ? (_Float16)src[idx] : (_Float16)0.0f;
}

// ---------------------------------------------------------------------------
// C(MxN) = sigmoid( A(MxK) @ W(NxK)^T )   via V_WMMA_F32_16X16X32_F16
// Block tile: 128 x (64*NSUB); wave tile: 64 x (16*NSUB).
// Async global->LDS staging (ASYNCcnt) with double-buffered LDS,
// one barrier per K-step, loads for step k+1 overlap WMMAs of step k.
// ---------------------------------------------------------------------------
template <int NSUB>
__global__ __launch_bounds__(256)
void gemm_sig(const _Float16* __restrict__ A, const _Float16* __restrict__ W,
              void* __restrict__ Out, int M, int N, int K, int out_f32) {
    (void)M;
    constexpr int BNt = 64 * NSUB;
    __shared__ _Float16 As[2][BM * LDT];
    __shared__ _Float16 Ws[2][BNt * LDT];

    const int t    = threadIdx.x;
    const int wave = t >> 5;
    const int lane = t & 31;
    const int wm   = wave & 1;   // which 64-row half of the M tile
    const int wn   = wave >> 1;  // which (16*NSUB)-col strip of the N tile
    const int bm   = blockIdx.y * BM;
    const int bn   = blockIdx.x * BNt;

    const int lrow = lane & 15;
    const int lhi  = (lane >> 4) & 1;
    const int r4   = t >> 2;     // 0..63
    const int c4   = t & 3;      // which 8-half chunk of the 32-half row

    // issue async global->LDS loads for one K-step into buffer `buf`
    auto issue = [&](int buf, int kb) {
        {
            unsigned la = (unsigned)(size_t)&As[buf][r4 * LDT + c4 * 8];
            unsigned long long ga =
                (unsigned long long)(size_t)&A[(size_t)(bm + r4) * K + kb + c4 * 8];
            asm volatile("global_load_async_to_lds_b128 %0, %1, off"
                         :: "v"(la), "v"(ga) : "memory");
            la = (unsigned)(size_t)&As[buf][(r4 + 64) * LDT + c4 * 8];
            ga = (unsigned long long)(size_t)&A[(size_t)(bm + r4 + 64) * K + kb + c4 * 8];
            asm volatile("global_load_async_to_lds_b128 %0, %1, off"
                         :: "v"(la), "v"(ga) : "memory");
        }
        #pragma unroll
        for (int w = 0; w < NSUB; ++w) {
            unsigned la = (unsigned)(size_t)&Ws[buf][(r4 + w * 64) * LDT + c4 * 8];
            unsigned long long ga =
                (unsigned long long)(size_t)&W[(size_t)(bn + r4 + w * 64) * K + kb + c4 * 8];
            asm volatile("global_load_async_to_lds_b128 %0, %1, off"
                         :: "v"(la), "v"(ga) : "memory");
        }
    };

    v8f acc[4][NSUB] = {};

    issue(0, 0);
    int buf = 0;
    for (int kb = 0; kb < K; kb += BK) {
        asm volatile("s_wait_asynccnt 0x0" ::: "memory");
        __syncthreads();
        if (kb + BK < K) issue(buf ^ 1, kb + BK);

        // B fragments: 32x16, elems 0-7 -> K=lhi*8+0..7, elems 8-15 -> +16
        v16h bfrag[NSUB];
        #pragma unroll
        for (int ns = 0; ns < NSUB; ++ns) {
            const _Float16* wp =
                &Ws[buf][(wn * 16 * NSUB + ns * 16 + lrow) * LDT + lhi * 8];
            v8h blo = *(const v8h*)wp;
            v8h bhi = *(const v8h*)(wp + 16);
            bfrag[ns] = __builtin_shufflevector(blo, bhi,
                0,1,2,3,4,5,6,7,8,9,10,11,12,13,14,15);
        }
        // A fragments: elems 0-7 -> K=lhi*8+0..7, elems 8-15 -> +16
        #pragma unroll
        for (int mt = 0; mt < 4; ++mt) {
            const _Float16* ap =
                &As[buf][(wm * 64 + mt * 16 + lrow) * LDT + lhi * 8];
            v8h alo = *(const v8h*)ap;
            v8h ahi = *(const v8h*)(ap + 16);
            v16h afrag = __builtin_shufflevector(alo, ahi,
                0,1,2,3,4,5,6,7,8,9,10,11,12,13,14,15);
            #pragma unroll
            for (int ns = 0; ns < NSUB; ++ns)
                acc[mt][ns] = __builtin_amdgcn_wmma_f32_16x16x32_f16(
                    false, afrag, false, bfrag[ns], (short)0,
                    acc[mt][ns], false, false);
        }
        buf ^= 1;
    }

    // epilogue: D layout lane->col (lane&15), VGPR v->row v + lhi*8
    const int colbase = bn + wn * 16 * NSUB;
    #pragma unroll
    for (int mt = 0; mt < 4; ++mt) {
        const int rbase = bm + wm * 64 + mt * 16 + lhi * 8;
        #pragma unroll
        for (int ns = 0; ns < NSUB; ++ns) {
            const int col = colbase + ns * 16 + lrow;
            #pragma unroll
            for (int v = 0; v < 8; ++v) {
                float s = 1.0f / (1.0f + __expf(-acc[mt][ns][v]));
                size_t idx = (size_t)(rbase + v) * N + col;
                if (out_f32) ((float*)Out)[idx] = s;
                else         ((_Float16*)Out)[idx] = (_Float16)s;
            }
        }
    }
}

// ---------------------------------------------------------------------------
// sample[:, i] = (x[:, i] >= u[i, :]) ? 1 : -1
// ---------------------------------------------------------------------------
__global__ void update_col(_Float16* __restrict__ sampleH, const float* __restrict__ x,
                           const float* __restrict__ u, int i) {
    int b = blockIdx.x * blockDim.x + threadIdx.x;
    float xv = x[(size_t)b * NSPINS + i];
    float uv = u[(size_t)i * BATCH + b];
    sampleH[(size_t)b * NSPINS + i] = (_Float16)((xv >= uv) ? 1.0f : -1.0f);
}

// ---------------------------------------------------------------------------
extern "C" void kernel_launch(void* const* d_in, const int* in_sizes, int n_in,
                              void* d_out, int out_size, void* d_ws, size_t ws_size,
                              hipStream_t stream) {
    (void)in_sizes; (void)n_in; (void)out_size; (void)ws_size;

    const float* sample = (const float*)d_in[0];
    const float* u      = (const float*)d_in[1];
    const float* W1     = (const float*)d_in[2];
    const float* W2     = (const float*)d_in[3];
    const float* W3     = (const float*)d_in[4];
    const float* W4     = (const float*)d_in[5];
    const float* W5     = (const float*)d_in[6];
    const float* W6     = (const float*)d_in[7];

    // workspace carve-up (256-B aligned slices)
    char* p = (char*)d_ws;
    auto take = [&](size_t bytes) {
        char* q = p;
        p += (bytes + 255) & ~(size_t)255;
        return q;
    };
    _Float16* W1h  = (_Float16*)take((size_t)HCH * NSPINS * 2);
    _Float16* W2h  = (_Float16*)take((size_t)HCH * HCH * 2);
    _Float16* W3h  = (_Float16*)take((size_t)HCH * HCH * 2);
    _Float16* W4h  = (_Float16*)take((size_t)HCH * HCH * 2);
    _Float16* W5h  = (_Float16*)take((size_t)HCH * HCH * 2);
    _Float16* W6h  = (_Float16*)take((size_t)NSPINS * HCH * 2);
    _Float16* smpH = (_Float16*)take((size_t)BATCH * NSPINS * 2);
    _Float16* actA = (_Float16*)take((size_t)BATCH * HCH * 2);
    _Float16* actB = (_Float16*)take((size_t)BATCH * HCH * 2);
    float*    x6   = (float*)   take((size_t)BATCH * NSPINS * 4);

    auto cvt = [&](const float* s, _Float16* d, int rows, int cols, int mode) {
        long long e = (long long)rows * cols;
        mask_cvt<<<(unsigned)((e + 255) / 256), 256, 0, stream>>>(s, d, rows, cols, mode);
    };
    // bake masks into f16 weights; convert sample
    cvt(W1, W1h, HCH, NSPINS, 0);
    cvt(W2, W2h, HCH, HCH, 1);
    cvt(W3, W3h, HCH, HCH, 1);
    cvt(W4, W4h, HCH, HCH, 1);
    cvt(W5, W5h, HCH, HCH, 1);
    cvt(W6, W6h, NSPINS, HCH, 2);
    cvt(sample, smpH, BATCH, NSPINS, 3);

    const dim3 gh(HCH / 128, BATCH / BM);   // hidden layers: 128x128 block tiles
    const dim3 g6(1, BATCH / BM);           // fc6: 128x64 block tiles

    auto net = [&](void* lastOut) {
        gemm_sig<2><<<gh, 256, 0, stream>>>(smpH, W1h, actA, BATCH, HCH, NSPINS, 0);
        gemm_sig<2><<<gh, 256, 0, stream>>>(actA, W2h, actB, BATCH, HCH, HCH, 0);
        gemm_sig<2><<<gh, 256, 0, stream>>>(actB, W3h, actA, BATCH, HCH, HCH, 0);
        gemm_sig<2><<<gh, 256, 0, stream>>>(actA, W4h, actB, BATCH, HCH, HCH, 0);
        gemm_sig<2><<<gh, 256, 0, stream>>>(actB, W5h, actA, BATCH, HCH, HCH, 0);
        gemm_sig<1><<<g6, 256, 0, stream>>>(actA, W6h, lastOut, BATCH, NSPINS, HCH, 1);
    };

    // 64 autoregressive sampling steps (full recompute each, like the reference)
    for (int i = 0; i < NSPINS; ++i) {
        net(x6);
        update_col<<<BATCH / 256, 256, 0, stream>>>(smpH, x6, u, i);
    }
    // final evaluation straight into d_out (f32, BATCH x NSPINS)
    net(d_out);
}
